// GradientTransform_13151189860739
// MI455X (gfx1250) — compile-verified, matrix operand order checked by reading, not run on hardware
//
#include <hip/hip_runtime.h>
#include <hip/hip_bf16.h>

typedef __attribute__((ext_vector_type(16))) __bf16 bf16x16;
typedef __attribute__((ext_vector_type(8)))  float  f32x8;

#define NORM_EPS 1e-7f

// ---------------------------------------------------------------------------
// Stage 1: row mask (any(u_row != 0) && any(v_row != 0))
// ---------------------------------------------------------------------------
__global__ void row_mask_kernel(const float* __restrict__ u,
                                const float* __restrict__ v,
                                float* __restrict__ mask, int XD, int DD) {
  const int row = blockIdx.x;
  const float* ur = u + (size_t)row * XD;
  const float* vr = v + (size_t)row * DD;
  int au = 0, av = 0;
  for (int i = threadIdx.x; i < XD; i += blockDim.x) au |= (ur[i] != 0.f);
  for (int i = threadIdx.x; i < DD; i += blockDim.x) av |= (vr[i] != 0.f);
  __shared__ int f0, f1;
  if (threadIdx.x == 0) { f0 = 0; f1 = 0; }
  __syncthreads();
  if (au) atomicOr(&f0, 1);
  if (av) atomicOr(&f1, 1);
  __syncthreads();
  if (threadIdx.x == 0) mask[row] = (f0 && f1) ? 1.f : 0.f;
}

// Deterministic k = sum(mask), single block.
__global__ void ksum_kernel(const float* __restrict__ mask,
                            float* __restrict__ kout, int n) {
  __shared__ float s[1024];
  float a = 0.f;
  for (int i = threadIdx.x; i < n; i += 1024) a += mask[i];
  s[threadIdx.x] = a;
  __syncthreads();
  for (int d = 512; d > 0; d >>= 1) {
    if (threadIdx.x < (unsigned)d) s[threadIdx.x] += s[threadIdx.x + d];
    __syncthreads();
  }
  if (threadIdx.x == 0) kout[0] = s[0];
}

// ---------------------------------------------------------------------------
// Stage 2: per-column sum(w*x), sum(w*x^2) in fixed-order chunks (deterministic)
// ---------------------------------------------------------------------------
__global__ void col_stats_partial(const float* __restrict__ x,
                                  const float* __restrict__ mask,
                                  float* __restrict__ p1, float* __restrict__ p2,
                                  int C, int rows_per_chunk) {
  const int col = blockIdx.x * 256 + threadIdx.x;
  const int chunk = blockIdx.y;
  const int r0 = chunk * rows_per_chunk;
  float s1 = 0.f, s2 = 0.f;
  for (int r = r0; r < r0 + rows_per_chunk; ++r) {
    const float w = mask[r];
    const float xv = x[(size_t)r * C + col];
    const float xw = xv * w;
    s1 += xw;
    s2 += xw * xv;
  }
  p1[(size_t)chunk * C + col] = s1;
  p2[(size_t)chunk * C + col] = s2;
}

__global__ void col_stats_finalize(const float* __restrict__ p1,
                                   const float* __restrict__ p2,
                                   const float* __restrict__ kptr,
                                   float* __restrict__ mean,
                                   float* __restrict__ inv,
                                   int C, int nchunk) {
  const int col = blockIdx.x * 256 + threadIdx.x;
  if (col >= C) return;
  float s1 = 0.f, s2 = 0.f;
  for (int c = 0; c < nchunk; ++c) {
    s1 += p1[(size_t)c * C + col];
    s2 += p2[(size_t)c * C + col];
  }
  const float k = kptr[0];
  const float m = s1 / k;
  float s = s2 - k * m * m;            // == sum(w*(x-m)^2)
  s = fmaxf(s, 0.f);
  const float sd = sqrtf(s / (k - 1.f));
  mean[col] = m;
  inv[col] = 1.f / (sd + NORM_EPS);
}

// ---------------------------------------------------------------------------
// Stage 3: normalize + convert to bf16 ; weight transpose-convert to bf16 [N,K]
// ---------------------------------------------------------------------------
__global__ void normalize_kernel(const float* __restrict__ x,
                                 const float* __restrict__ mean,
                                 const float* __restrict__ inv,
                                 __bf16* __restrict__ y, int C) {
  const size_t idx = (size_t)blockIdx.x * 256 + threadIdx.x;
  const int col = (int)(idx % C);
  y[idx] = (__bf16)((x[idx] - mean[col]) * inv[col]);
}

// W is [K,N] row-major f32 -> Wt is [N,K] row-major bf16
__global__ void transpose_bf16_kernel(const float* __restrict__ W,
                                      __bf16* __restrict__ Wt, int K, int N) {
  const size_t idx = (size_t)blockIdx.x * 256 + threadIdx.x;
  const int n = (int)(idx % N);
  const int k = (int)(idx / N);
  Wt[(size_t)n * K + k] = (__bf16)W[idx];
}

// ---------------------------------------------------------------------------
// Stage 4: GEMM  out = act(A[M,K] @ Bt[N,K]^T + bias)   (bf16 in, f32 acc)
//   HIDDEN=true : relu, store bf16
//   HIDDEN=false: * mask[row], store f32
// Block: 256 thr (8 waves), tile 128x128, K-step 32, LDS double-buffered.
// Wave tile 32x64 = 2x4 WMMA 16x16 tiles. M,N % 128 == 0, K % 32 == 0 here.
// ---------------------------------------------------------------------------
template<bool HIDDEN>
__global__ __launch_bounds__(256)
void gemm_wmma_kernel(const __bf16* __restrict__ A,
                      const __bf16* __restrict__ Bt,
                      const float* __restrict__ bias,
                      const float* __restrict__ mask,
                      void* __restrict__ outp,
                      int M, int N, int K) {
  __shared__ __align__(16) __bf16 As[2][128][40];  // +8 pad, 80B pitch
  __shared__ __align__(16) __bf16 Bs[2][128][40];

  const int n0 = blockIdx.x * 128;
  const int m0 = blockIdx.y * 128;
  const int tid = threadIdx.x;
  const int lane = tid & 31;
  const int wave = tid >> 5;
  const int wr = (wave & 3) * 32;   // 4 waves tile M
  const int wn = (wave >> 2) * 64;  // 2 waves tile N
  const int half = lane >> 4;       // lane half selects K sub-chunk
  const int r = lane & 15;

  // Each thread copies exactly 2x16B from A and 2x16B from Bt per K-tile.
  const int rowL = tid >> 2;        // 0..63 (second row is rowL+64)
  const int kp   = (tid & 3) * 8;   // bf16 element offset within 32-wide K tile
  const __bf16* gA0 = A  + (size_t)(m0 + rowL)      * K + kp;
  const __bf16* gA1 = A  + (size_t)(m0 + rowL + 64) * K + kp;
  const __bf16* gB0 = Bt + (size_t)(n0 + rowL)      * K + kp;
  const __bf16* gB1 = Bt + (size_t)(n0 + rowL + 64) * K + kp;

  // preload tile 0 into buffer 0
  uint4 ra0 = *(const uint4*)gA0;
  uint4 ra1 = *(const uint4*)gA1;
  uint4 rb0 = *(const uint4*)gB0;
  uint4 rb1 = *(const uint4*)gB1;
  *(uint4*)&As[0][rowL][kp]      = ra0;
  *(uint4*)&As[0][rowL + 64][kp] = ra1;
  *(uint4*)&Bs[0][rowL][kp]      = rb0;
  *(uint4*)&Bs[0][rowL + 64][kp] = rb1;
  __syncthreads();

  f32x8 acc[2][4] = {};
  int cur = 0;

  for (int k0 = 0; k0 < K; k0 += 32) {
    const int nxt = k0 + 32;
    const bool more = nxt < K;        // uniform branch
    if (more) {
      ra0 = *(const uint4*)(gA0 + nxt);
      ra1 = *(const uint4*)(gA1 + nxt);
      rb0 = *(const uint4*)(gB0 + nxt);
      rb1 = *(const uint4*)(gB1 + nxt);
      __builtin_prefetch(gA0 + nxt + 32, 0, 0);   // two tiles ahead
      __builtin_prefetch(gB0 + nxt + 32, 0, 0);
    }

    // Fragment layout (16-bit A 16x32): lane half=0 -> K {0..7,16..23},
    // half=1 -> K {8..15,24..31}; two ds_load_b128 per fragment.
    union Frag { bf16x16 v; uint4 q[2]; };
    Frag fa[2], fb[4];
#pragma unroll
    for (int fm = 0; fm < 2; ++fm) {
      const __bf16* p = &As[cur][wr + fm * 16 + r][half * 8];
      fa[fm].q[0] = *(const uint4*)p;
      fa[fm].q[1] = *(const uint4*)(p + 16);
    }
#pragma unroll
    for (int fn = 0; fn < 4; ++fn) {
      const __bf16* p = &Bs[cur][wn + fn * 16 + r][half * 8];
      fb[fn].q[0] = *(const uint4*)p;
      fb[fn].q[1] = *(const uint4*)(p + 16);
    }
#pragma unroll
    for (int fm = 0; fm < 2; ++fm)
#pragma unroll
      for (int fn = 0; fn < 4; ++fn)
        acc[fm][fn] = __builtin_amdgcn_wmma_f32_16x16x32_bf16(
            false, fa[fm].v, false, fb[fn].v, (short)0, acc[fm][fn],
            false, false);

    if (more) {
      const int nb = cur ^ 1;
      *(uint4*)&As[nb][rowL][kp]      = ra0;
      *(uint4*)&As[nb][rowL + 64][kp] = ra1;
      *(uint4*)&Bs[nb][rowL][kp]      = rb0;
      *(uint4*)&Bs[nb][rowL + 64][kp] = rb1;
    }
    __syncthreads();
    cur ^= 1;
  }

  // Epilogue. C/D layout: VGPR g holds row (g + 8*half), col = lane&15.
#pragma unroll
  for (int fm = 0; fm < 2; ++fm) {
    const int rowb = m0 + wr + fm * 16 + half * 8;
    float mv[8];
    if (!HIDDEN) {
#pragma unroll
      for (int g = 0; g < 8; ++g) mv[g] = mask[rowb + g];
    }
#pragma unroll
    for (int fn = 0; fn < 4; ++fn) {
      const int col = n0 + wn + fn * 16 + r;
      const float bv = bias[col];
#pragma unroll
      for (int g = 0; g < 8; ++g) {
        const int row = rowb + g;
        float val = acc[fm][fn][g] + bv;
        if (HIDDEN) {
          val = fmaxf(val, 0.f);
          ((__bf16*)outp)[(size_t)row * N + col] = (__bf16)val;
        } else {
          ((float*)outp)[(size_t)row * N + col] = val * mv[g];
        }
      }
    }
  }
}

// ---------------------------------------------------------------------------
extern "C" void kernel_launch(void* const* d_in, const int* in_sizes, int n_in,
                              void* d_out, int out_size, void* d_ws, size_t ws_size,
                              hipStream_t stream) {
  const float* u   = (const float*)d_in[0];
  const float* v   = (const float*)d_in[1];
  const float* uW1 = (const float*)d_in[2];
  const float* ub1 = (const float*)d_in[3];
  const float* uW2 = (const float*)d_in[4];
  const float* ub2 = (const float*)d_in[5];
  const float* vW1 = (const float*)d_in[6];
  const float* vb1 = (const float*)d_in[7];
  const float* vW2 = (const float*)d_in[8];
  const float* vb2 = (const float*)d_in[9];

  const int XD = in_sizes[5];        // 1024
  const int XH = in_sizes[3];        // 2048
  const int DD = in_sizes[9];        // 4096
  const int DH = in_sizes[7];        // 8192
  const int NR = in_sizes[0] / XD;   // 16384
  const int NCHUNK = 32;
  const int RPC = NR / NCHUNK;

  // ---- workspace layout (256B aligned slices) ----
  char* base = (char*)d_ws;
  size_t off = 0;
  auto take = [&](size_t bytes) -> void* {
    void* p = base + off;
    off = (off + bytes + 255) & ~(size_t)255;
    return p;
  };
  float*  mask   = (float*)take((size_t)NR * 4);
  float*  kval   = (float*)take(4);
  float*  u_p1   = (float*)take((size_t)NCHUNK * XD * 4);
  float*  u_p2   = (float*)take((size_t)NCHUNK * XD * 4);
  float*  v_p1   = (float*)take((size_t)NCHUNK * DD * 4);
  float*  v_p2   = (float*)take((size_t)NCHUNK * DD * 4);
  float*  u_mean = (float*)take((size_t)XD * 4);
  float*  u_inv  = (float*)take((size_t)XD * 4);
  float*  v_mean = (float*)take((size_t)DD * 4);
  float*  v_inv  = (float*)take((size_t)DD * 4);
  __bf16* u_nrm  = (__bf16*)take((size_t)NR * XD * 2);
  __bf16* v_nrm  = (__bf16*)take((size_t)NR * DD * 2);
  __bf16* uW1t   = (__bf16*)take((size_t)XH * XD * 2);
  __bf16* uW2t   = (__bf16*)take((size_t)XD * XH * 2);
  __bf16* vW1t   = (__bf16*)take((size_t)DH * DD * 2);
  __bf16* vW2t   = (__bf16*)take((size_t)DD * DH * 2);
  __bf16* h_u    = (__bf16*)take((size_t)NR * XH * 2);
  __bf16* h_v    = (__bf16*)take((size_t)NR * DH * 2);

  // ---- stats / normalization ----
  row_mask_kernel<<<NR, 256, 0, stream>>>(u, v, mask, XD, DD);
  ksum_kernel<<<1, 1024, 0, stream>>>(mask, kval, NR);

  col_stats_partial<<<dim3(XD / 256, NCHUNK), 256, 0, stream>>>(u, mask, u_p1, u_p2, XD, RPC);
  col_stats_partial<<<dim3(DD / 256, NCHUNK), 256, 0, stream>>>(v, mask, v_p1, v_p2, DD, RPC);
  col_stats_finalize<<<XD / 256, 256, 0, stream>>>(u_p1, u_p2, kval, u_mean, u_inv, XD, NCHUNK);
  col_stats_finalize<<<DD / 256, 256, 0, stream>>>(v_p1, v_p2, kval, v_mean, v_inv, DD, NCHUNK);

  normalize_kernel<<<(unsigned)(((size_t)NR * XD) / 256), 256, 0, stream>>>(u, u_mean, u_inv, u_nrm, XD);
  normalize_kernel<<<(unsigned)(((size_t)NR * DD) / 256), 256, 0, stream>>>(v, v_mean, v_inv, v_nrm, DD);

  // ---- weights -> bf16 [N,K] ----
  transpose_bf16_kernel<<<(unsigned)(((size_t)XD * XH) / 256), 256, 0, stream>>>(uW1, uW1t, XD, XH);
  transpose_bf16_kernel<<<(unsigned)(((size_t)XH * XD) / 256), 256, 0, stream>>>(uW2, uW2t, XH, XD);
  transpose_bf16_kernel<<<(unsigned)(((size_t)DD * DH) / 256), 256, 0, stream>>>(vW1, vW1t, DD, DH);
  transpose_bf16_kernel<<<(unsigned)(((size_t)DH * DD) / 256), 256, 0, stream>>>(vW2, vW2t, DH, DD);

  // ---- GEMMs (WMMA bf16 -> f32 acc) ----
  float* out1 = (float*)d_out;
  float* out2 = out1 + (size_t)NR * XD;

  gemm_wmma_kernel<true ><<<dim3(XH / 128, NR / 128), 256, 0, stream>>>(u_nrm, uW1t, ub1, mask, h_u, NR, XH, XD);
  gemm_wmma_kernel<false><<<dim3(XD / 128, NR / 128), 256, 0, stream>>>(h_u, uW2t, ub2, mask, out1, NR, XD, XH);
  gemm_wmma_kernel<true ><<<dim3(DH / 128, NR / 128), 256, 0, stream>>>(v_nrm, vW1t, vb1, mask, h_v, NR, DH, DD);
  gemm_wmma_kernel<false><<<dim3(DD / 128, NR / 128), 256, 0, stream>>>(h_v, vW2t, vb2, mask, out2, NR, DD, DH);
}
